// MoETransformerMT_66116726554795
// MI455X (gfx1250) — matrix operand order repaired and checked by reading, not compiled, not run on hardware
//
#include <hip/hip_runtime.h>
#include <cmath>

// ---------------------------------------------------------------------------
// CDNA5 / gfx1250 MoE-Transformer forward.
// GEMMs: v_wmma_f32_16x16x32_bf16, double-buffered LDS tiles; bf16 weight
// tiles are DMA'd by the Tensor Data Mover (tensor_load_to_lds + TENSORcnt)
// when the toolchain exposes the builtin.
// ---------------------------------------------------------------------------

typedef __attribute__((ext_vector_type(16))) __bf16 v16bf;
typedef __attribute__((ext_vector_type(8)))  float  v8f;
typedef __attribute__((ext_vector_type(4)))  unsigned int u32x4;
typedef __attribute__((ext_vector_type(8)))  int i32x8v;
typedef __attribute__((ext_vector_type(4)))  int i32x4v;

union BF16Frag { v16bf v; unsigned int u[8]; };

#if __has_builtin(__builtin_amdgcn_tensor_load_to_lds)
#define HAVE_TDM 1
#else
#define HAVE_TDM 0
#endif

__device__ __forceinline__ void wait_tensorcnt0() {
#if __has_builtin(__builtin_amdgcn_s_wait_tensorcnt)
  __builtin_amdgcn_s_wait_tensorcnt(0);
#else
  asm volatile("s_wait_tensorcnt 0x0" ::: "memory");
#endif
}

#if HAVE_TDM
// Build a 2D TDM descriptor: tile 32(K,contig) x 64(N rows) of 2-byte elems,
// row stride = ldb elems, LDS padding 2 dwords after every 16 dwords (matches
// the 18-dword LDS row pitch used by the WMMA fragment loader).
__device__ __forceinline__ void tdm_load_b_tile(const void* gsrc, unsigned lds_byte,
                                                unsigned remK, unsigned remN,
                                                unsigned long long strideElems) {
  unsigned long long ga = (unsigned long long)gsrc;
  u32x4 g0;
  g0.x = 1u;                                               // count=1 (valid D#)
  g0.y = lds_byte;                                         // lds_addr
  g0.z = (unsigned)(ga & 0xFFFFFFFFu);                     // global_addr[31:0]
  g0.w = (unsigned)((ga >> 32) & 0x01FFFFFFu) | (2u << 30);// addr[56:32] | type=2
  i32x8v g1;
  g1[0] = (int)((1u << 16)   // data_size = 2 bytes
              | (1u << 20)   // pad_enable
              | (3u << 22)   // pad_interval: 16 dwords
              | (1u << 25)); // pad_amount: 2 dwords
  g1[1] = (int)((remK & 0xFFFFu) << 16);                   // tensor_dim0 lo16
  g1[2] = (int)((remK >> 16) | ((remN & 0xFFFFu) << 16));  // dim0 hi | dim1 lo
  g1[3] = (int)((remN >> 16) | (32u << 16));               // dim1 hi | tile_dim0=32
  g1[4] = (int)64;                                         // tile_dim1=64, tile_dim2=0
  g1[5] = (int)(unsigned)(strideElems & 0xFFFFFFFFu);      // dim0_stride[31:0]
  g1[6] = (int)(unsigned)((strideElems >> 32) & 0xFFFFu);  // dim0_stride[47:32]
  g1[7] = 0;
  i32x4v z4 = {0, 0, 0, 0};
#if __clang_major__ >= 23
  i32x8v z8 = {0, 0, 0, 0, 0, 0, 0, 0};
  __builtin_amdgcn_tensor_load_to_lds(g0, g1, z4, z4, z8, 0);
#else
  __builtin_amdgcn_tensor_load_to_lds(g0, g1, z4, z4, 0);
#endif
}
#else
__device__ __forceinline__ void tdm_load_b_tile(const void*, unsigned, unsigned,
                                                unsigned, unsigned long long) {}
#endif

__device__ __forceinline__ unsigned pack_bf16(float a, float b) {
  unsigned ua = __float_as_uint(a), ub = __float_as_uint(b);
  ua += 0x7FFFu + ((ua >> 16) & 1u);           // round-to-nearest-even
  ub += 0x7FFFu + ((ub >> 16) & 1u);
  return (ua >> 16) | (ub & 0xFFFF0000u);
}

__device__ __forceinline__ uint2 ld4_bf16pair(const float* p) {
  const float4 f = *(const float4*)p;
  return make_uint2(pack_bf16(f.x, f.y), pack_bf16(f.z, f.w));
}
__device__ __forceinline__ uint2 ld4_bf16pair(const unsigned short* p) {
  return *(const uint2*)p;                      // already bf16 bits, pair-packed
}
__device__ __forceinline__ unsigned pack2(float a, float b) { return pack_bf16(a, b); }
__device__ __forceinline__ unsigned pack2(unsigned short a, unsigned short b) {
  return (unsigned)a | ((unsigned)b << 16);
}

// ---------------------------------------------------------------------------
// Batched WMMA GEMM:  C[z] (opt +=) = act(A[z] f32 [M,K]) x B[z] (+bias)(+relu)
//   TRANSB=1: B is [N,K] row-major (weights / K-head for scores)
//   TRANSB=0: B is [K,N] row-major (MoE w1/w2 / V-head for attn*V)
//   rowScale != null:  C[m,n] += rowScale[m] * (acc+bias)   (MoE top-2 combine)
// Block tile 128x64x32, 256 threads = 8 waves, wave tile 32x32 (2x2 WMMA).
// Double-buffered LDS; bf16 TRANSB tiles go through the TDM when available.
// Requires M%128==0, N%64==0, K%32==0 (true for every GEMM in this model).
// ---------------------------------------------------------------------------
#define BM 128
#define BN 64
#define BK 32
#define ASTR 18   // dwords per LDS A row (16 + 2 pad)
#define BSTR 18
#define A_BUF_DW (BM * ASTR)                              // 2304 dwords per A buffer
#define B_BUF_DW (BN * BSTR)                              // 1152 dwords per B buffer
#define GEMM_SMEM ((2 * A_BUF_DW + 2 * B_BUF_DW) * 4)     // 27648 bytes

struct GemmP {
  const float* A; const void* B; float* C;
  const float* bias; const float* rowScale;
  int M, N, K, lda, ldb, ldc, H, relu;
  long long bsA, hsA, bsB, hsB, bsC, hsC;   // per-batch(b) / per-head(h) strides
};

template <typename TB, bool TRANSB>
__global__ __launch_bounds__(256) void k_gemm_wmma(GemmP p) {
  extern __shared__ unsigned int smem[];    // dynamic LDS starts at offset 0
  // layout (dwords): A0[A_BUF_DW] A1[A_BUF_DW] B0[B_BUF_DW] B1[B_BUF_DW]
  // NOTE: select buffers by integer offset only (no pointer arrays into LDS —
  // constant aggregates of addrspacecast'd LDS pointers break ld.lld).

  const int tid  = threadIdx.x;
  const int lane = tid & 31;
  const int wid  = tid >> 5;
  const int wm   = wid & 3;          // 4 wave-rows of 32
  const int wn   = wid >> 2;         // 2 wave-cols of 32
  const int z = blockIdx.z, b = z / p.H, h = z % p.H;

  const float* Ab = p.A + b * p.bsA + h * p.hsA + (long long)blockIdx.y * BM * p.lda;
  const TB*    Bb = (const TB*)p.B + b * p.bsB + h * p.hsB;
  float*       Cb = p.C + b * p.bsC + h * p.hsC;
  const int n0 = blockIdx.x * BN;

  const bool useTDM = HAVE_TDM && TRANSB && (sizeof(TB) == 2);

  v8f acc[2][2];
  { v8f zz = {}; for (int s = 0; s < 2; ++s) for (int t = 0; t < 2; ++t) acc[s][t] = zz; }

  uint2 aReg[4];
  uint2 bReg[2];
  unsigned bPk[4];

  auto loadA = [&](int kb) {
#pragma unroll
    for (int t = 0; t < 4; ++t) {
      int g = tid + t * 256;
      int row = g >> 3, c4 = (g & 7) << 2;
      const float* ap = Ab + (long long)row * p.lda + kb + c4;
      aReg[t] = ld4_bf16pair(ap);
      if (kb + BK < p.K) __builtin_prefetch(ap + BK, 0, 1);   // global_prefetch_b8
    }
  };
  auto storeA = [&](int buf) {
    unsigned int* dst = smem + buf * A_BUF_DW;
#pragma unroll
    for (int t = 0; t < 4; ++t) {
      int g = tid + t * 256;
      int row = g >> 3, c4 = (g & 7) << 2;
      *(uint2*)&dst[row * ASTR + (c4 >> 1)] = aReg[t];
    }
  };
  auto loadB = [&](int kb) {
    if (TRANSB) {
#pragma unroll
      for (int t = 0; t < 2; ++t) {
        int g = tid + t * 256;
        int n = g >> 3, c4 = (g & 7) << 2;
        bReg[t] = ld4_bf16pair(Bb + (long long)(n0 + n) * p.ldb + kb + c4);
      }
    } else {
      int n = tid & 63, kbase = (tid >> 6) << 3;
#pragma unroll
      for (int kk = 0; kk < 8; kk += 2) {
        int k = kbase + kk;
        TB e0 = Bb[(long long)(kb + k) * p.ldb + n0 + n];
        TB e1 = Bb[(long long)(kb + k + 1) * p.ldb + n0 + n];
        bPk[kk >> 1] = pack2(e0, e1);
      }
    }
  };
  auto storeB = [&](int buf) {
    unsigned int* dst = smem + 2 * A_BUF_DW + buf * B_BUF_DW;
    if (TRANSB) {
#pragma unroll
      for (int t = 0; t < 2; ++t) {
        int g = tid + t * 256;
        int n = g >> 3, c4 = (g & 7) << 2;
        *(uint2*)&dst[n * BSTR + (c4 >> 1)] = bReg[t];
      }
    } else {
      int n = tid & 63, kbase = (tid >> 6) << 3;
#pragma unroll
      for (int kk = 0; kk < 8; kk += 2)
        dst[n * BSTR + ((kbase + kk) >> 1)] = bPk[kk >> 1];
    }
  };
  auto issueTDM = [&](int kb, int buf) {
    if (wid == 0) {
      const TB* src = Bb + (long long)n0 * p.ldb + kb;
      tdm_load_b_tile(src,
                      (unsigned)(2 * A_BUF_DW * 4) + (unsigned)buf * (B_BUF_DW * 4u),
                      (unsigned)(p.K - kb), (unsigned)(p.N - n0),
                      (unsigned long long)p.ldb);
    }
  };

  // ---- pipeline prologue: stage tile 0 into buffer 0 ----
  loadA(0);
  storeA(0);
  if (useTDM) issueTDM(0, 0);
  else { loadB(0); storeB(0); }

  const int nkt = p.K / BK;
  for (int kt = 0; kt < nkt; ++kt) {
    if (useTDM && wid == 0) wait_tensorcnt0();
    __syncthreads();                       // tile[kt] visible; buffer nxt free
    const int cur = kt & 1, nxt = cur ^ 1;
    const int kbn = (kt + 1) * BK;
    const bool more = kbn < p.K;
    if (more) {
      loadA(kbn);                          // global loads overlap WMMA below
      if (useTDM) issueTDM(kbn, nxt);      // TDM DMA overlaps WMMA below
      else loadB(kbn);
    }

    // ---- fragments per documented CDNA5 16-bit WMMA layouts ----
    const int mh = lane >> 4, l15 = lane & 15;
    const unsigned int* A_ = smem + cur * A_BUF_DW;
    const unsigned int* B_ = smem + 2 * A_BUF_DW + cur * B_BUF_DW;
    BF16Frag af[2], bf[2];
#pragma unroll
    for (int s = 0; s < 2; ++s) {
      int ml = wm * 32 + s * 16 + l15;
#pragma unroll
      for (int v = 0; v < 8; ++v) {
        int k = ((v & 4) << 2) + ((v & 3) << 1) + (mh << 3);   // A 16x32 layout
        af[s].u[v] = A_[ml * ASTR + (k >> 1)];
      }
    }
#pragma unroll
    for (int t = 0; t < 2; ++t) {
      int nl = wn * 32 + t * 16 + l15;
#pragma unroll
      for (int v = 0; v < 8; ++v)                              // B 32x16 layout
        bf[t].u[v] = B_[nl * BSTR + ((v & 4) << 1) + (v & 3) + (mh << 2)];
    }
#pragma unroll
    for (int s = 0; s < 2; ++s)
#pragma unroll
      for (int t = 0; t < 2; ++t)
        acc[s][t] = __builtin_amdgcn_wmma_f32_16x16x32_bf16(
            false, af[s].v, false, bf[t].v, (short)0, acc[s][t], false, false);

    if (more) {
      storeA(nxt);
      if (!useTDM) storeB(nxt);
    }
  }

  // ---- epilogue: bias / relu / row-scaled accumulate ----
  const int mh = lane >> 4, l15 = lane & 15;
#pragma unroll
  for (int s = 0; s < 2; ++s) {
#pragma unroll
    for (int t = 0; t < 2; ++t) {
      int nG = n0 + wn * 32 + t * 16 + l15;
      float bv = p.bias ? p.bias[nG] : 0.0f;
#pragma unroll
      for (int r = 0; r < 8; ++r) {
        int mG = blockIdx.y * BM + wm * 32 + s * 16 + mh * 8 + r;
        float v = acc[s][t][r] + bv;
        if (p.relu) v = v > 0.0f ? v : 0.0f;
        float* cp = Cb + (long long)mG * p.ldc + nG;
        if (p.rowScale) *cp += p.rowScale[mG] * v;
        else            *cp = v;
      }
    }
  }
}

// ---------------------------------------------------------------------------
// Helper kernels (wave32-native)
// ---------------------------------------------------------------------------
__global__ void k_f32_to_bf16(const float* __restrict__ in,
                              unsigned short* __restrict__ out, long long n) {
  long long i = (long long)blockIdx.x * blockDim.x + threadIdx.x;
  long long st = (long long)gridDim.x * blockDim.x;
  for (; i < n; i += st) {
    unsigned u = __float_as_uint(in[i]);
    u += 0x7FFFu + ((u >> 16) & 1u);
    out[i] = (unsigned short)(u >> 16);
  }
}

__global__ void k_zero(float* p, long long n) {
  long long i = (long long)blockIdx.x * blockDim.x + threadIdx.x;
  long long st = (long long)gridDim.x * blockDim.x;
  for (; i < n; i += st) p[i] = 0.0f;
}

__global__ void k_embed(const int* __restrict__ ids, const float* __restrict__ emb,
                        const float* __restrict__ pe, float* __restrict__ out,
                        int S, int D, float scale, long long total) {
  long long i = (long long)blockIdx.x * blockDim.x + threadIdx.x;
  long long st = (long long)gridDim.x * blockDim.x;
  for (; i < total; i += st) {
    long long tok = i / D; int d = (int)(i % D); int s = (int)(tok % S);
    out[i] = emb[(long long)ids[tok] * D + d] * scale + pe[(long long)s * D + d];
  }
}

// y = LN(x (+ r)) * g + b     one wave32 per 512-wide row, 8 waves/block
__global__ __launch_bounds__(256) void k_ln(const float* __restrict__ x,
                                            const float* __restrict__ r,
                                            const float* __restrict__ g,
                                            const float* __restrict__ b,
                                            float* __restrict__ y) {
  const int wid = threadIdx.x >> 5, lane = threadIdx.x & 31;
  const long long row = (long long)blockIdx.x * 8 + wid;
  const float* xp = x + row * 512;
  const float* rp = r ? r + row * 512 : nullptr;
  float v[16]; float s = 0.0f;
#pragma unroll
  for (int i = 0; i < 4; ++i) {
    int j = i * 128 + lane * 4;
    float4 a = *(const float4*)(xp + j);
    if (rp) { float4 c = *(const float4*)(rp + j); a.x += c.x; a.y += c.y; a.z += c.z; a.w += c.w; }
    v[i*4+0]=a.x; v[i*4+1]=a.y; v[i*4+2]=a.z; v[i*4+3]=a.w;
    s += a.x + a.y + a.z + a.w;
  }
  for (int o = 16; o; o >>= 1) s += __shfl_xor(s, o, 32);
  float mu = s * (1.0f / 512.0f);
  float var = 0.0f;
#pragma unroll
  for (int e = 0; e < 16; ++e) { float d = v[e] - mu; var += d * d; }
  for (int o = 16; o; o >>= 1) var += __shfl_xor(var, o, 32);
  float rstd = rsqrtf(var * (1.0f / 512.0f) + 1e-5f);
#pragma unroll
  for (int i = 0; i < 4; ++i) {
    int j = i * 128 + lane * 4;
    float4 gg = *(const float4*)(g + j);
    float4 bb = *(const float4*)(b + j);
    float4 o4 = make_float4((v[i*4+0]-mu)*rstd*gg.x+bb.x, (v[i*4+1]-mu)*rstd*gg.y+bb.y,
                            (v[i*4+2]-mu)*rstd*gg.z+bb.z, (v[i*4+3]-mu)*rstd*gg.w+bb.w);
    *(float4*)(y + row * 512 + j) = o4;
  }
}

// softmax over 512-wide score rows; fused 1/sqrt(HD) scale and causal mask
__global__ __launch_bounds__(256) void k_softmax(float* __restrict__ sc, int Sq,
                                                 float scale, int causal) {
  const int wid = threadIdx.x >> 5, lane = threadIdx.x & 31;
  const long long row = (long long)blockIdx.x * 8 + wid;
  const int qi = (int)(row % Sq);
  float* p = sc + row * 512;
  float v[16]; float mx = -3.4e38f;
#pragma unroll
  for (int i = 0; i < 4; ++i) {
    int j0 = i * 128 + lane * 4;
    float4 f = *(const float4*)(p + j0);
    float t[4] = {f.x, f.y, f.z, f.w};
#pragma unroll
    for (int c = 0; c < 4; ++c) {
      float xv = t[c] * scale;
      if (causal && (j0 + c) > qi) xv = -1e9f;
      v[i*4+c] = xv;
      mx = fmaxf(mx, xv);
    }
  }
  for (int o = 16; o; o >>= 1) mx = fmaxf(mx, __shfl_xor(mx, o, 32));
  float sum = 0.0f;
#pragma unroll
  for (int e = 0; e < 16; ++e) { v[e] = __expf(v[e] - mx); sum += v[e]; }
  for (int o = 16; o; o >>= 1) sum += __shfl_xor(sum, o, 32);
  float inv = 1.0f / sum;
#pragma unroll
  for (int i = 0; i < 4; ++i) {
    int j0 = i * 128 + lane * 4;
    *(float4*)(p + j0) = make_float4(v[i*4]*inv, v[i*4+1]*inv, v[i*4+2]*inv, v[i*4+3]*inv);
  }
}

// top-2 router: combineT[e*M+tok] = gate prob if e in top2 else 0; ep += mean gate
__global__ __launch_bounds__(256) void k_gate(const float* __restrict__ x,
                                              const float* __restrict__ rw,
                                              const float* __restrict__ rb,
                                              float* __restrict__ combineT,
                                              float* __restrict__ ep, int Mtok) {
  const int wid = threadIdx.x >> 5, lane = threadIdx.x & 31;
  const int tok = blockIdx.x * 8 + wid;
  const float* xp = x + (long long)tok * 512;
  float lg[8];
#pragma unroll
  for (int e = 0; e < 8; ++e) {
    const float* wp = rw + e * 512;
    float s = 0.0f;
    for (int d = lane; d < 512; d += 32) s += xp[d] * wp[d];
    for (int o = 16; o; o >>= 1) s += __shfl_xor(s, o, 32);
    lg[e] = s + rb[e];
  }
  if (lane == 0) {
    float mx = lg[0];
    for (int e = 1; e < 8; ++e) mx = fmaxf(mx, lg[e]);
    float pb[8], sum = 0.0f;
    for (int e = 0; e < 8; ++e) { pb[e] = __expf(lg[e] - mx); sum += pb[e]; }
    float inv = 1.0f / sum;
    for (int e = 0; e < 8; ++e) pb[e] *= inv;
    int i1 = 0;
    for (int e = 1; e < 8; ++e) if (pb[e] > pb[i1]) i1 = e;
    int i2 = -1;
    for (int e = 0; e < 8; ++e) if (e != i1 && (i2 < 0 || pb[e] > pb[i2])) i2 = e;
    float invM = 1.0f / (float)Mtok;
    for (int e = 0; e < 8; ++e) {
      combineT[(long long)e * Mtok + tok] = (e == i1 || e == i2) ? pb[e] : 0.0f;
      atomicAdd(&ep[e], pb[e] * invM);
    }
  }
}

__global__ void k_aux(const float* __restrict__ ep, float* __restrict__ aux) {
  if (threadIdx.x == 0 && blockIdx.x == 0) {
    float s = 0.0f;
    for (int e = 0; e < 8; ++e) s += ep[e] * logf(ep[e] + 1e-9f);
    aux[0] += s;
  }
}

// ---------------------------------------------------------------------------
// Host-side orchestration
// ---------------------------------------------------------------------------
static void launch_gemm(hipStream_t s, bool transB, bool b_bf16,
                        const float* A, const void* B, float* C,
                        const float* bias, const float* rowScale,
                        int M, int N, int K, int lda, int ldb, int ldc, int relu,
                        int Z = 1, int H = 1,
                        long long bsA = 0, long long hsA = 0,
                        long long bsB = 0, long long hsB = 0,
                        long long bsC = 0, long long hsC = 0) {
  GemmP p{A, B, C, bias, rowScale, M, N, K, lda, ldb, ldc, H, relu,
          bsA, hsA, bsB, hsB, bsC, hsC};
  dim3 grid(N / BN, M / BM, Z), blk(256);
  if (transB) {
    if (b_bf16) k_gemm_wmma<unsigned short, true><<<grid, blk, GEMM_SMEM, s>>>(p);
    else        k_gemm_wmma<float,          true><<<grid, blk, GEMM_SMEM, s>>>(p);
  } else {
    if (b_bf16) k_gemm_wmma<unsigned short, false><<<grid, blk, GEMM_SMEM, s>>>(p);
    else        k_gemm_wmma<float,          false><<<grid, blk, GEMM_SMEM, s>>>(p);
  }
}

extern "C" void kernel_launch(void* const* d_in, const int* in_sizes, int n_in,
                              void* d_out, int out_size, void* d_ws, size_t ws_size,
                              hipStream_t stream) {
  (void)in_sizes; (void)n_in; (void)ws_size;
  const int D = 512, Hh = 8, F = 2048, S = 512, Bn = 4, V = 32000;
  const int M = Bn * S;                       // 2048 tokens per stream

  // ---- pytree leaf indices (jax sorted-key flatten order) ----
  // params.dec[L]: ca(in_b,in_w,out_b,out_w) ln1(b,g) ln2(b,g) ln3(b,g)
  //                moe(b1,b2,rb,rw,w1,w2) sa(in_b,in_w,out_b,out_w)   -> 20 leaves
  // params.enc[L]: attn(4) ln1(2) ln2(2) moe(6)                       -> 14 leaves
  const int ENC0 = 60;
  const int LN_DEC_B = 102, LN_DEC_G = 103, LN_ENC_B = 104, LN_ENC_G = 105;
  const int OUT_B = 106, OUT_W = 107, SRC_EMB = 108, TGT_EMB = 109;
  const int PE = 110, SRC_IDS = 111, TGT_IDS = 112;
  auto f32p = [&](int i) { return (const float*)d_in[i]; };
  auto i32p = [&](int i) { return (const int*)d_in[i]; };

  // ---- workspace carve ----
  char* wsp = (char*)d_ws; size_t off = 0;
  auto carve = [&](size_t bytes) { void* p = wsp + off; off = (off + bytes + 255) & ~(size_t)255; return p; };

  unsigned short *enc_inw[3], *enc_outw[3], *sa_inw[3], *sa_outw[3], *ca_inw[3], *ca_outw[3];
  unsigned short *w1bf[6], *w2bf[6];
  for (int L = 0; L < 3; ++L) {
    enc_inw[L]  = (unsigned short*)carve((size_t)1536 * 512 * 2);
    enc_outw[L] = (unsigned short*)carve((size_t)512 * 512 * 2);
    sa_inw[L]   = (unsigned short*)carve((size_t)1536 * 512 * 2);
    sa_outw[L]  = (unsigned short*)carve((size_t)512 * 512 * 2);
    ca_inw[L]   = (unsigned short*)carve((size_t)1536 * 512 * 2);
    ca_outw[L]  = (unsigned short*)carve((size_t)512 * 512 * 2);
  }
  for (int i = 0; i < 6; ++i) {
    w1bf[i] = (unsigned short*)carve((size_t)8 * 512 * 2048 * 2);
    w2bf[i] = (unsigned short*)carve((size_t)8 * 2048 * 512 * 2);
  }
  unsigned short* outw_bf = (unsigned short*)carve((size_t)V * 512 * 2);

  float* x_enc   = (float*)carve((size_t)M * D * 4);
  float* x_dec   = (float*)carve((size_t)M * D * 4);
  float* tmp     = (float*)carve((size_t)M * D * 4);
  float* ctx     = (float*)carve((size_t)M * D * 4);
  float* moe_acc = (float*)carve((size_t)M * D * 4);
  float* qkv     = (float*)carve((size_t)M * 3 * D * 4);   // also q(512) + kv(1024)
  float* scores  = (float*)carve((size_t)Bn * Hh * S * S * 4);
  float* moe_h   = (float*)carve((size_t)M * F * 4);
  float* combT   = (float*)carve((size_t)8 * M * 4);
  float* ep      = (float*)carve(256);
  float* qkv2    = qkv + (long long)M * 512;               // kv region for cross-attn

  auto conv = [&](const float* src, unsigned short* dst, long long n) {
    long long blocks = (n + 255) / 256; if (blocks > 8192) blocks = 8192;
    k_f32_to_bf16<<<dim3((unsigned)blocks), dim3(256), 0, stream>>>(src, dst, n);
  };

  // ---- one-time weight f32 -> bf16 conversion ----
  for (int L = 0; L < 3; ++L) {
    int eb = ENC0 + L * 14, db = L * 20;
    conv(f32p(eb + 1),  enc_inw[L],  1536LL * 512);
    conv(f32p(eb + 3),  enc_outw[L], 512LL * 512);
    conv(f32p(eb + 12), w1bf[L],     8LL * 512 * 2048);
    conv(f32p(eb + 13), w2bf[L],     8LL * 2048 * 512);
    conv(f32p(db + 17), sa_inw[L],   1536LL * 512);
    conv(f32p(db + 19), sa_outw[L],  512LL * 512);
    conv(f32p(db + 1),  ca_inw[L],   1536LL * 512);
    conv(f32p(db + 3),  ca_outw[L],  512LL * 512);
    conv(f32p(db + 14), w1bf[3 + L], 8LL * 512 * 2048);
    conv(f32p(db + 15), w2bf[3 + L], 8LL * 2048 * 512);
  }
  conv(f32p(OUT_W), outw_bf, (long long)V * 512);

  // ---- aux-loss accumulator ----
  float* aux_out = (float*)d_out + (out_size - 1);
  k_zero<<<1, 32, 0, stream>>>(aux_out, 1);

  // ---- embeddings + positional encoding ----
  const float escale = sqrtf(512.0f);
  k_embed<<<4096, 256, 0, stream>>>(i32p(SRC_IDS), f32p(SRC_EMB), f32p(PE), x_enc, S, D, escale, (long long)M * D);
  k_embed<<<4096, 256, 0, stream>>>(i32p(TGT_IDS), f32p(TGT_EMB), f32p(PE), x_dec, S, D, escale, (long long)M * D);

  // ---- shared sub-blocks ----
  auto moe_block = [&](float* xs, const float* b1, const float* b2,
                       const float* rbp, const float* rwp,
                       unsigned short* w1, unsigned short* w2,
                       const float* lg, const float* lb) {
    k_zero<<<1, 32, 0, stream>>>(ep, 8);
    k_gate<<<M / 8, 256, 0, stream>>>(xs, rwp, rbp, combT, ep, M);
    k_aux<<<1, 32, 0, stream>>>(ep, aux_out);
    k_zero<<<2048, 256, 0, stream>>>(moe_acc, (long long)M * D);
    for (int e = 0; e < 8; ++e) {
      launch_gemm(stream, false, true, xs, w1 + (long long)e * 512 * 2048, moe_h,
                  b1 + e * 2048, nullptr, M, 2048, 512, 512, 2048, 2048, /*relu*/1);
      launch_gemm(stream, false, true, moe_h, w2 + (long long)e * 2048 * 512, moe_acc,
                  b2 + e * 512, combT + (long long)e * M, M, 512, 2048, 2048, 512, 512, 0);
    }
    k_ln<<<M / 8, 256, 0, stream>>>(xs, moe_acc, lg, lb, xs);
  };

  auto self_attn = [&](float* xs, unsigned short* iw, const float* ib,
                       unsigned short* ow, const float* ob,
                       const float* lg, const float* lb, int causal) {
    launch_gemm(stream, true, true, xs, iw, qkv, ib, nullptr, M, 1536, 512, 512, 512, 1536, 0);
    launch_gemm(stream, true, false, qkv, qkv + 512, scores, nullptr, nullptr,
                S, S, 64, 1536, 1536, S, 0, Bn * Hh, Hh,
                (long long)S * 1536, 64, (long long)S * 1536, 64,
                (long long)Hh * S * S, (long long)S * S);
    k_softmax<<<Bn * Hh * S / 8, 256, 0, stream>>>(scores, S, 0.125f, causal);
    launch_gemm(stream, false, false, scores, qkv + 1024, ctx, nullptr, nullptr,
                S, 64, S, S, 1536, 512, 0, Bn * Hh, Hh,
                (long long)Hh * S * S, (long long)S * S,
                (long long)S * 1536, 64, (long long)S * 512, 64);
    launch_gemm(stream, true, true, ctx, ow, tmp, ob, nullptr, M, 512, 512, 512, 512, 512, 0);
    k_ln<<<M / 8, 256, 0, stream>>>(xs, tmp, lg, lb, xs);
  };

  // ---- encoder ----
  for (int L = 0; L < 3; ++L) {
    int eb = ENC0 + L * 14;
    self_attn(x_enc, enc_inw[L], f32p(eb + 0), enc_outw[L], f32p(eb + 2),
              f32p(eb + 5), f32p(eb + 4), /*causal*/0);
    moe_block(x_enc, f32p(eb + 8), f32p(eb + 9), f32p(eb + 10), f32p(eb + 11),
              w1bf[L], w2bf[L], f32p(eb + 7), f32p(eb + 6));
  }
  k_ln<<<M / 8, 256, 0, stream>>>(x_enc, nullptr, f32p(LN_ENC_G), f32p(LN_ENC_B), x_enc);

  // ---- decoder ----
  for (int L = 0; L < 3; ++L) {
    int db = L * 20;
    // causal self-attention
    self_attn(x_dec, sa_inw[L], f32p(db + 16), sa_outw[L], f32p(db + 18),
              f32p(db + 5), f32p(db + 4), /*causal*/1);
    // cross-attention: q from x_dec, k/v from encoder memory
    launch_gemm(stream, true, true, x_dec, ca_inw[L], qkv, f32p(db + 0), nullptr,
                M, 512, 512, 512, 512, 512, 0);
    launch_gemm(stream, true, true, x_enc, ca_inw[L] + 512 * 512, qkv2,
                f32p(db + 0) + 512, nullptr, M, 1024, 512, 512, 512, 1024, 0);
    launch_gemm(stream, true, false, qkv, qkv2, scores, nullptr, nullptr,
                S, S, 64, 512, 1024, S, 0, Bn * Hh, Hh,
                (long long)S * 512, 64, (long long)S * 1024, 64,
                (long long)Hh * S * S, (long long)S * S);
    k_softmax<<<Bn * Hh * S / 8, 256, 0, stream>>>(scores, S, 0.125f, 0);
    launch_gemm(stream, false, false, scores, qkv2 + 512, ctx, nullptr, nullptr,
                S, 64, S, S, 1024, 512, 0, Bn * Hh, Hh,
                (long long)Hh * S * S, (long long)S * S,
                (long long)S * 1024, 64, (long long)S * 512, 64);
    launch_gemm(stream, true, true, ctx, ca_outw[L], tmp, f32p(db + 2), nullptr,
                M, 512, 512, 512, 512, 512, 0);
    k_ln<<<M / 8, 256, 0, stream>>>(x_dec, tmp, f32p(db + 7), f32p(db + 6), x_dec);
    // MoE
    moe_block(x_dec, f32p(db + 10), f32p(db + 11), f32p(db + 12), f32p(db + 13),
              w1bf[3 + L], w2bf[3 + L], f32p(db + 9), f32p(db + 8));
  }
  k_ln<<<M / 8, 256, 0, stream>>>(x_dec, nullptr, f32p(LN_DEC_G), f32p(LN_DEC_B), x_dec);

  // ---- output projection -> logits ----
  launch_gemm(stream, true, true, x_dec, outw_bf, (float*)d_out, f32p(OUT_B), nullptr,
              M, V, 512, 512, 512, V, 0);
}
